// DKMIL_23974507446304
// MI455X (gfx1250) — matrix operand order repaired and checked by reading, not compiled, not run on hardware
//
#include <hip/hip_runtime.h>
#include <cstddef>

// ---------------- CDNA5 WMMA types ----------------
typedef __attribute__((ext_vector_type(16))) __bf16        v16bf;
typedef __attribute__((ext_vector_type(8)))  float         v8f;
typedef __attribute__((ext_vector_type(4)))  unsigned int  v4u;

union Frag { v4u u[2]; v16bf bf; };

__device__ __forceinline__ float leaky1(float x) { return x > 0.f ? x : 0.01f * x; }

// f32 -> bf16 round-to-nearest-even
__device__ __forceinline__ unsigned short f2bf(float f) {
  unsigned u = __float_as_uint(f);
  u += 0x7FFFu + ((u >> 16) & 1u);
  return (unsigned short)(u >> 16);
}

// pack 8 consecutive f32 (two float4) into 8 bf16 in a v4u
__device__ __forceinline__ v4u pack8(float4 a, float4 b) {
  v4u r;
  r[0] = (unsigned)f2bf(a.x) | ((unsigned)f2bf(a.y) << 16);
  r[1] = (unsigned)f2bf(a.z) | ((unsigned)f2bf(a.w) << 16);
  r[2] = (unsigned)f2bf(b.x) | ((unsigned)f2bf(b.y) << 16);
  r[3] = (unsigned)f2bf(b.z) | ((unsigned)f2bf(b.w) << 16);
  return r;
}

#define BM   128
#define BN   128
#define BKK  32
#define LSTR 40   // bf16 elements per LDS row (80B => every fragment 16B-aligned)

// ============================================================
// Generic WMMA GEMM (templated on layout): C = epilogue(A * B)
//   TA: A stored [K x M] (use A[k*lda+m]);   else [M x K]
//   TB: compute A @ Bm^T with Bm [N x K];    else B [K x N]
//   act: 0 none, 1 leaky, 2 tanh, 3 euclid (sqrt(max(rowv+colv-2acc,0)))
//   C = act(acc + bias[col]) * postScale + residual
// ============================================================
template<int TA, int TB>
__global__ __launch_bounds__(256) void wmma_gemm_t(
    const float* __restrict__ A, const float* __restrict__ B, float* __restrict__ C,
    int M, int N, int K, int lda, int ldb, int ldc,
    const float* __restrict__ bias, const float* __restrict__ rowv,
    const float* __restrict__ colv, const float* __restrict__ residual,
    int act, float postScale)
{
  __shared__ alignas(16) unsigned short As[BM * LSTR];
  __shared__ alignas(16) unsigned short Bs[BN * LSTR];

  const int tid   = threadIdx.x;
  const int lane  = tid & 31;
  const int wave  = tid >> 5;
  const int waveM = wave & 3;   // 4 waves along M (32 rows each)
  const int waveN = wave >> 2;  // 2 waves along N (64 cols each)
  const int m0 = blockIdx.y * BM;
  const int n0 = blockIdx.x * BN;

  v8f acc[2][4];
  for (int i = 0; i < 2; ++i)
    for (int j = 0; j < 4; ++j)
      for (int r = 0; r < 8; ++r) acc[i][j][r] = 0.f;

  const bool mFull = (m0 + BM <= M);
  const bool nFull = (n0 + BN <= N);

  const int kt = (K + BKK - 1) / BKK;
  for (int t = 0; t < kt; ++t) {
    const int k0 = t * BKK;
    const bool kFull = (k0 + BKK <= K);

    // ---------------- stage A tile (BM x BKK) as bf16, K-major ----------------
    if (TA == 0) {
      if (kFull && mFull) {
        const int r    = tid >> 1;
        const int half = tid & 1;
        const float* src = A + (size_t)(m0 + r) * lda + k0 + half * 16;
        const float4* s4 = (const float4*)src;
        float4 f0 = s4[0], f1 = s4[1], f2 = s4[2], f3 = s4[3];
        unsigned short* dst = &As[r * LSTR + half * 16];
        ((v4u*)dst)[0] = pack8(f0, f1);
        ((v4u*)(dst + 8))[0] = pack8(f2, f3);
        if (t + 1 < kt) __builtin_prefetch(src + BKK, 0, 1);
      } else {
        for (int i = 0; i < 16; ++i) {
          int e = i * 256 + tid;
          int r = e >> 5, k = e & 31;
          int gm = m0 + r, gk = k0 + k;
          float v = (gm < M && gk < K) ? A[(size_t)gm * lda + gk] : 0.f;
          As[r * LSTR + k] = f2bf(v);
        }
      }
    } else { // TA == 1 : A is [K x M], stage transposed
      if (kFull && mFull) {
        const int k  = tid >> 3;
        const int mb = (tid & 7) * 16;
        const float* src = B ? (A + (size_t)(k0 + k) * lda + m0 + mb)
                             : nullptr; // (B always non-null; keeps optimizer honest)
        src = A + (size_t)(k0 + k) * lda + m0 + mb;
        const float4* s4 = (const float4*)src;
        float4 f[4] = { s4[0], s4[1], s4[2], s4[3] };
        const float* fv = (const float*)f;
        for (int j = 0; j < 16; ++j)
          As[(mb + j) * LSTR + k] = f2bf(fv[j]);
        if (t + 1 < kt) __builtin_prefetch(src + (size_t)BKK * lda, 0, 1);
      } else {
        for (int i = 0; i < 16; ++i) {
          int e = i * 256 + tid;
          int r = e & 127, k = e >> 7;
          int gm = m0 + r, gk = k0 + k;
          float v = (gm < M && gk < K) ? A[(size_t)gk * lda + gm] : 0.f;
          As[r * LSTR + k] = f2bf(v);
        }
      }
    }

    // ---------------- stage B tile as LDS[n][k] ----------------
    if (TB == 1) { // Bm [N x K] -> direct row copy
      if (kFull && nFull) {
        const int r    = tid >> 1;
        const int half = tid & 1;
        const float* src = B + (size_t)(n0 + r) * ldb + k0 + half * 16;
        const float4* s4 = (const float4*)src;
        float4 f0 = s4[0], f1 = s4[1], f2 = s4[2], f3 = s4[3];
        unsigned short* dst = &Bs[r * LSTR + half * 16];
        ((v4u*)dst)[0] = pack8(f0, f1);
        ((v4u*)(dst + 8))[0] = pack8(f2, f3);
        if (t + 1 < kt) __builtin_prefetch(src + BKK, 0, 1);
      } else {
        for (int i = 0; i < 16; ++i) {
          int e = i * 256 + tid;
          int n = e >> 5, k = e & 31;
          int gn = n0 + n, gk = k0 + k;
          float v = (gn < N && gk < K) ? B[(size_t)gn * ldb + gk] : 0.f;
          Bs[n * LSTR + k] = f2bf(v);
        }
      }
    } else { // B [K x N] -> transpose into LDS
      if (kFull && nFull) {
        const int k  = tid >> 3;
        const int nb = (tid & 7) * 16;
        const float* src = B + (size_t)(k0 + k) * ldb + n0 + nb;
        const float4* s4 = (const float4*)src;
        float4 f[4] = { s4[0], s4[1], s4[2], s4[3] };
        const float* fv = (const float*)f;
        for (int j = 0; j < 16; ++j)
          Bs[(nb + j) * LSTR + k] = f2bf(fv[j]);
        if (t + 1 < kt) __builtin_prefetch(src + (size_t)BKK * ldb, 0, 1);
      } else {
        for (int i = 0; i < 16; ++i) {
          int e = i * 256 + tid;
          int n = e & 127, k = e >> 7;
          int gn = n0 + n, gk = k0 + k;
          float v = (gn < N && gk < K) ? B[(size_t)gk * ldb + gn] : 0.f;
          Bs[n * LSTR + k] = f2bf(v);
        }
      }
    }
    __syncthreads();

    // ---- fragment loads (ds_load_b128 pairs) per ISA 16-bit layouts
    Frag a[2], b[4];
    const int rsel = lane & 15;
    const int kh = (lane < 16) ? 0 : 8;    // A: K halves {0..7|16..23} vs {8..15|24..31}
    const int kb = (lane < 16) ? 0 : 8;    // B: K halves {0..7|16..23} vs {8..15|24..31}
    for (int i = 0; i < 2; ++i) {
      int row = waveM * 32 + i * 16 + rsel;
      a[i].u[0] = *(const v4u*)&As[row * LSTR + kh];
      a[i].u[1] = *(const v4u*)&As[row * LSTR + 16 + kh];
    }
    for (int j = 0; j < 4; ++j) {
      int col = waveN * 64 + j * 16 + rsel;
      b[j].u[0] = *(const v4u*)&Bs[col * LSTR + kb];
      b[j].u[1] = *(const v4u*)&Bs[col * LSTR + kb + 16];
    }
    for (int i = 0; i < 2; ++i)
      for (int j = 0; j < 4; ++j)
        acc[i][j] = __builtin_amdgcn_wmma_f32_16x16x32_bf16(
            false, a[i].bf, false, b[j].bf, (short)0, acc[i][j], false, false);
    __syncthreads();
  }

  // ---- epilogue (C/D layout: lane n = lane&15, vgpr r -> m = r + (lane<16?0:8))
  const int nLoc = lane & 15;
  const int mOff = (lane < 16) ? 0 : 8;
  for (int i = 0; i < 2; ++i)
    for (int j = 0; j < 4; ++j) {
      int col = n0 + waveN * 64 + j * 16 + nLoc;
      if (col >= N) continue;
      for (int r = 0; r < 8; ++r) {
        int row = m0 + waveM * 32 + i * 16 + mOff + r;
        if (row >= M) continue;
        float v = acc[i][j][r];
        if (act == 3) {
          v = rowv[row] + colv[col] - 2.f * v;
          v = sqrtf(fmaxf(v, 0.f));
        } else {
          if (bias) v += bias[col];
          if (act == 1) v = leaky1(v);
          else if (act == 2) v = tanhf(v);
        }
        v *= postScale;
        if (residual) v += residual[(size_t)row * ldc + col];
        C[(size_t)row * ldc + col] = v;
      }
    }
}

// ============================================================
// conv1 (1->20, 5x5, 28->24) + leaky + maxpool2  => [n,20,12,12]
// ============================================================
__global__ void conv1_pool(const float* __restrict__ x, const float* __restrict__ w,
                           const float* __restrict__ b, float* __restrict__ out, int total)
{
  int t = blockIdx.x * blockDim.x + threadIdx.x;
  if (t >= total) return;
  int ox = t % 12, oy = (t / 12) % 12, c = (t / 144) % 20, n = t / 2880;
  const float* xi = x + (size_t)n * 784;
  const float* wc = w + c * 25;
  float bb = b[c];
  float mx = -3.4e38f;
  for (int dy = 0; dy < 2; ++dy)
    for (int dx = 0; dx < 2; ++dx) {
      int cy = 2 * oy + dy, cx = 2 * ox + dx;
      float s = bb;
      for (int i = 0; i < 5; ++i)
        for (int j = 0; j < 5; ++j)
          s += xi[(cy + i) * 28 + (cx + j)] * wc[i * 5 + j];
      mx = fmaxf(mx, leaky1(s));
    }
  out[t] = mx;
}

// conv2 (20->50, 5x5, 12->8) + leaky + maxpool2 => [n,50,4,4] == [n,800]
__global__ void conv2_pool(const float* __restrict__ p1, const float* __restrict__ w,
                           const float* __restrict__ b, float* __restrict__ out, int total)
{
  int t = blockIdx.x * blockDim.x + threadIdx.x;
  if (t >= total) return;
  int ox = t % 4, oy = (t / 4) % 4, c = (t / 16) % 50, n = t / 800;
  const float* pi = p1 + (size_t)n * 2880;
  float bb = b[c];
  float mx = -3.4e38f;
  for (int dy = 0; dy < 2; ++dy)
    for (int dx = 0; dx < 2; ++dx) {
      int cy = 2 * oy + dy, cx = 2 * ox + dx;
      float s = bb;
      for (int ic = 0; ic < 20; ++ic) {
        const float* pc = pi + ic * 144;
        const float* wc = w + c * 500 + ic * 25;
        for (int i = 0; i < 5; ++i)
          for (int j = 0; j < 5; ++j)
            s += pc[(cy + i) * 12 + (cx + j)] * wc[i * 5 + j];
      }
      mx = fmaxf(mx, leaky1(s));
    }
  out[t] = mx;
}

// softmax over axis 0 (columns) of X [R x C], one thread per column
__global__ void softmax_col_k(float* __restrict__ X, int R, int C)
{
  int col = blockIdx.x * blockDim.x + threadIdx.x;
  if (col >= C) return;
  float m = -3.4e38f;
  for (int r = 0; r < R; ++r) m = fmaxf(m, X[(size_t)r * C + col]);
  float s = 0.f;
  for (int r = 0; r < R; ++r) s += expf(X[(size_t)r * C + col] - m);
  float inv = 1.f / s;
  for (int r = 0; r < R; ++r) {
    size_t o = (size_t)r * C + col;
    X[o] = expf(X[o] - m) * inv;
  }
}

// softmax over a length-n vector, single block of 256
__global__ void softmax_vec_k(const float* __restrict__ in, float* __restrict__ out, int n)
{
  __shared__ float red[256];
  int t = threadIdx.x;
  float m = -3.4e38f;
  for (int i = t; i < n; i += 256) m = fmaxf(m, in[i]);
  red[t] = m; __syncthreads();
  for (int o = 128; o > 0; o >>= 1) { if (t < o) red[t] = fmaxf(red[t], red[t + o]); __syncthreads(); }
  m = red[0]; __syncthreads();
  float s = 0.f;
  for (int i = t; i < n; i += 256) s += expf(in[i] - m);
  red[t] = s; __syncthreads();
  for (int o = 128; o > 0; o >>= 1) { if (t < o) red[t] += red[t + o]; __syncthreads(); }
  s = red[0];
  float inv = 1.f / s;
  for (int i = t; i < n; i += 256) out[i] = expf(in[i] - m) * inv;
}

// deterministic top-S (descending, stable ties -> lower index), single block
__global__ void topk_k(const float* __restrict__ scores, int* __restrict__ idx, int n, int S)
{
  __shared__ float sv[2048];
  __shared__ float rv[256];
  __shared__ int   ri[256];
  int t = threadIdx.x;
  for (int i = t; i < n; i += 256) sv[i] = scores[i];
  __syncthreads();
  for (int s = 0; s < S; ++s) {
    float bv = -3.4e38f; int bi = n;
    for (int i = t; i < n; i += 256) {
      float v = sv[i];
      if (v > bv || (v == bv && i < bi)) { bv = v; bi = i; }
    }
    rv[t] = bv; ri[t] = bi; __syncthreads();
    for (int o = 128; o > 0; o >>= 1) {
      if (t < o) {
        if (rv[t + o] > rv[t] || (rv[t + o] == rv[t] && ri[t + o] < ri[t])) {
          rv[t] = rv[t + o]; ri[t] = ri[t + o];
        }
      }
      __syncthreads();
    }
    if (t == 0) { idx[s] = ri[0]; sv[ri[0]] = -3.4e38f; }
    __syncthreads();
  }
}

// B_bag: per-row softmax over gathered columns aff_s[row, idx[0..S)]
__global__ void bag_softmax_k(const float* __restrict__ affs, const int* __restrict__ idx,
                              float* __restrict__ bag, int S)
{
  __shared__ float red[256];
  __shared__ float vv[256];
  int row = blockIdx.x, t = threadIdx.x;
  float v = -3.4e38f;
  if (t < S) v = affs[(size_t)row * 2048 + idx[t]];
  vv[t] = v;
  red[t] = v; __syncthreads();
  for (int o = 128; o > 0; o >>= 1) { if (t < o) red[t] = fmaxf(red[t], red[t + o]); __syncthreads(); }
  float m = red[0]; __syncthreads();
  float e = (t < S) ? expf(vv[t] - m) : 0.f;
  red[t] = e; __syncthreads();
  for (int o = 128; o > 0; o >>= 1) { if (t < o) red[t] += red[t + o]; __syncthreads(); }
  float s = red[0];
  if (t < S) bag[(size_t)row * S + t] = e / s;
}

__global__ void rowsq_k(const float* __restrict__ X, float* __restrict__ out, int rows, int cols)
{
  int i = blockIdx.x * blockDim.x + threadIdx.x;
  if (i >= rows) return;
  float s = 0.f;
  const float* r = X + (size_t)i * cols;
  for (int k = 0; k < cols; ++k) s += r[k] * r[k];
  out[i] = s;
}

// s[j] = sum_i A3[i][j] * w4[i] + b4[0]   (A3 [Ki x n], no activation)
__global__ void colvec_dot_k(const float* __restrict__ A3, const float* __restrict__ w4,
                             const float* __restrict__ b4, float* __restrict__ out, int n, int Ki)
{
  int j = blockIdx.x * blockDim.x + threadIdx.x;
  if (j >= n) return;
  float s = b4[0];
  for (int i = 0; i < Ki; ++i) s += A3[(size_t)i * n + j] * w4[i];
  out[j] = s;
}

// a[i] = leaky( sum_k AV[i][k]*AU[i][k]*aw[k] + awb )
__global__ void gate_dot_k(const float* __restrict__ AV, const float* __restrict__ AU,
                           const float* __restrict__ aw, const float* __restrict__ awb,
                           float* __restrict__ out, int n, int Kd)
{
  int i = blockIdx.x * blockDim.x + threadIdx.x;
  if (i >= n) return;
  float s = awb[0];
  const float* av = AV + (size_t)i * Kd;
  const float* au = AU + (size_t)i * Kd;
  for (int k = 0; k < Kd; ++k) s += av[k] * au[k] * aw[k];
  out[i] = leaky1(s);
}

// out[j] = sum_n A[n] * H[n][j]  (single block, cols <= 256)
__global__ void wsum_k(const float* __restrict__ A, const float* __restrict__ H,
                       float* __restrict__ out, int n, int cols)
{
  int j = threadIdx.x;
  if (j >= cols) return;
  float s = 0.f;
  for (int i = 0; i < n; ++i) s += A[i] * H[(size_t)i * cols + j];
  out[j] = s;
}

// out[o] = leaky( dot(in, W[o]) + b[o] )
__global__ void small_linear_k(const float* __restrict__ in, const float* __restrict__ W,
                               const float* __restrict__ b, float* __restrict__ out, int O, int K)
{
  int o = blockIdx.x * blockDim.x + threadIdx.x;
  if (o >= O) return;
  float s = b[o];
  const float* w = W + (size_t)o * K;
  for (int k = 0; k < K; ++k) s += in[k] * w[k];
  out[o] = leaky1(s);
}

__global__ void concat_bag_k(const float* __restrict__ a, const float* __restrict__ bmat,
                             float* __restrict__ o, int n, int c1, int c2)
{
  int t = blockIdx.x * blockDim.x + threadIdx.x;
  int C = c1 + c2;
  if (t >= n * C) return;
  int row = t / C, col = t % C;
  o[t] = (col < c1) ? a[(size_t)row * c1 + col] : bmat[(size_t)row * c2 + (col - c1)];
}

__global__ void concat_vec_k(const float* __restrict__ a, const float* __restrict__ v,
                             float* __restrict__ o, int n, int c1, int c2)
{
  int t = blockIdx.x * blockDim.x + threadIdx.x;
  int C = c1 + c2;
  if (t >= n * C) return;
  int row = t / C, col = t % C;
  o[t] = (col < c1) ? a[(size_t)row * c1 + col] : v[col - c1];
}

__global__ void copy_k(const float* __restrict__ s, float* __restrict__ d, int n)
{
  int i = blockIdx.x * blockDim.x + threadIdx.x;
  if (i < n) d[i] = s[i];
}

__global__ void final_k(const float* __restrict__ bvec, const float* __restrict__ w,
                        const float* __restrict__ b, float* __restrict__ out)
{
  float s = b[0];
  for (int k = 0; k < 256; ++k) s += bvec[k] * w[k];
  float y = 1.f / (1.f + expf(-s));
  y = fminf(fmaxf(y, 1e-5f), 1.f - 1e-5f);
  out[0] = y;
}

// ============================================================
// Host side
// ============================================================
static inline int cdiv(int a, int b) { return (a + b - 1) / b; }
static constexpr float BN_SCALE_H = 0.9999950000374997f;  // 1/sqrt(1+1e-5)

static void launch_gemm(hipStream_t s, const float* A, const float* B, float* C,
                        int M, int N, int K, int lda, int ldb, int ldc, int tA, int tB,
                        const float* bias, const float* rowv, const float* colv,
                        const float* res, int act, float post)
{
  dim3 g(cdiv(N, BN), cdiv(M, BM)), b(256);
  if (tA == 0 && tB == 1)
    wmma_gemm_t<0, 1><<<g, b, 0, s>>>(A, B, C, M, N, K, lda, ldb, ldc,
                                      bias, rowv, colv, res, act, post);
  else if (tA == 0 && tB == 0)
    wmma_gemm_t<0, 0><<<g, b, 0, s>>>(A, B, C, M, N, K, lda, ldb, ldc,
                                      bias, rowv, colv, res, act, post);
  else
    wmma_gemm_t<1, 0><<<g, b, 0, s>>>(A, B, C, M, N, K, lda, ldb, ldc,
                                      bias, rowv, colv, res, act, post);
}

extern "C" void kernel_launch(void* const* d_in, const int* in_sizes, int n_in,
                              void* d_out, int out_size, void* d_ws, size_t ws_size,
                              hipStream_t stream)
{
  (void)in_sizes; (void)n_in; (void)out_size; (void)ws_size;
  const float* Bimg  = (const float*)d_in[0];
  const float* KBimg = (const float*)d_in[1];
  const float* c1w = (const float*)d_in[2];  const float* c1b = (const float*)d_in[3];
  const float* c2w = (const float*)d_in[4];  const float* c2b = (const float*)d_in[5];
  const float* embw = (const float*)d_in[6]; const float* embb = (const float*)d_in[7];
  const float* sk_w[3][3], *sk_b[3][3];
  for (int s = 0; s < 3; ++s)
    for (int l = 0; l < 3; ++l) {
      sk_w[s][l] = (const float*)d_in[8 + s * 6 + l * 2];
      sk_b[s][l] = (const float*)d_in[8 + s * 6 + l * 2 + 1];
    }
  const float* mk_w1 = (const float*)d_in[26]; const float* mk_b1 = (const float*)d_in[27];
  const float* mk_w2 = (const float*)d_in[28]; const float* mk_b2 = (const float*)d_in[29];
  const float* mk_w3 = (const float*)d_in[30]; const float* mk_b3 = (const float*)d_in[31];
  const float* mk_w4 = (const float*)d_in[32]; const float* mk_b4 = (const float*)d_in[33];
  const float* fuse_w = (const float*)d_in[34]; const float* fuse_b = (const float*)d_in[35];
  const float* a1_few = (const float*)d_in[36]; const float* a1_feb = (const float*)d_in[37];
  const float* a1_vw  = (const float*)d_in[38]; const float* a1_vb  = (const float*)d_in[39];
  const float* a1_uw  = (const float*)d_in[40]; const float* a1_ub  = (const float*)d_in[41];
  const float* a1_aww = (const float*)d_in[42]; const float* a1_awb = (const float*)d_in[43];
  const float* a1_ew  = (const float*)d_in[44]; const float* a1_eb  = (const float*)d_in[45];
  const float* a2_few = (const float*)d_in[46]; const float* a2_feb = (const float*)d_in[47];
  const float* a2_vw  = (const float*)d_in[48]; const float* a2_vb  = (const float*)d_in[49];
  const float* a2_uw  = (const float*)d_in[50]; const float* a2_ub  = (const float*)d_in[51];
  const float* a2_aww = (const float*)d_in[52]; const float* a2_awb = (const float*)d_in[53];
  const float* a2_ew  = (const float*)d_in[54]; const float* a2_eb  = (const float*)d_in[55];
  const float* semb_w = (const float*)d_in[56]; const float* semb_b = (const float*)d_in[57];
  const float* cls_w  = (const float*)d_in[58]; const float* cls_b  = (const float*)d_in[59];

  float* out = (float*)d_out;

  // ---- workspace bump allocator (floats)
  float* W = (float*)d_ws;
  size_t off = 0;
  auto alloc = [&](size_t nfl) { float* p = W + off; off += nfl; return p; };

  float* p1    = alloc(2048u * 20 * 144);   // conv1 pooled (reused for X3big later)
  float* p2    = alloc(2048u * 800);        // conv2 pooled (reused for concat later)
  float* Bdt   = alloc(2048u * 256);
  float* KBdt  = alloc(2048u * 256);
  float* X1b   = alloc(2048u * 128);
  float* X2b   = alloc(2048u * 128);
  float* X3s   = alloc(2048u * 256);
  float* feat  = alloc(2048u * 2048);
  float* aff   = alloc(2048u * 2048);
  float* affs  = alloc(2048u * 2048);
  float* Xs    = alloc(2048u * 256);
  float* Ks    = alloc(2048u * 256);
  float* xn    = alloc(2048);
  float* yn    = alloc(2048);
  float* Mmat  = alloc(128u * 128);
  float* a3    = alloc(128u * 2048);
  float* svec  = alloc(2048);
  float* scor  = alloc(2048);
  int*   idxb  = (int*)alloc(256);
  float* bag   = alloc(2048u * 204);
  float* Bfuse = alloc(2048u * 256);
  float* H1    = alloc(2048u * 256);
  float* avec  = alloc(2048);
  float* Avec  = alloc(2048);
  float* AH    = alloc(256);
  float* M1v   = alloc(256);
  float* X2cat = alloc(2048u * 512);
  float* H2    = alloc(2048u * 256);
  float* A2v   = alloc(2048);
  float* AH2   = alloc(256);
  float* m2    = alloc(512);
  float* bvec  = alloc(256);
  float* X3big = p1;   // 5.9M floats >= 4.2M; conv scratch dead by then
  float* catb  = p2;   // 1.64M floats >= 0.95M

  // ---- data_transform(B) and data_transform(KB)
  const int tot1 = 2048 * 20 * 144, tot2 = 2048 * 800;
  conv1_pool<<<cdiv(tot1, 256), 256, 0, stream>>>(Bimg, c1w, c1b, p1, tot1);
  conv2_pool<<<cdiv(tot2, 256), 256, 0, stream>>>(p1, c2w, c2b, p2, tot2);
  launch_gemm(stream, p2, embw, Bdt, 2048, 256, 800, 800, 800, 256, 0, 1,
              embb, nullptr, nullptr, nullptr, 1, 1.f);
  conv1_pool<<<cdiv(tot1, 256), 256, 0, stream>>>(KBimg, c1w, c1b, p1, tot1);
  conv2_pool<<<cdiv(tot2, 256), 256, 0, stream>>>(p1, c2w, c2b, p2, tot2);
  launch_gemm(stream, p2, embw, KBdt, 2048, 256, 800, 800, 800, 256, 0, 1,
              embb, nullptr, nullptr, nullptr, 1, 1.f);

  // ---- skip_block helper
  auto skip = [&](const float* Xin, int d, int si, float* X3buf, float* outbuf) {
    launch_gemm(stream, Xin, sk_w[si][0], X1b, 2048, 128, d, d, d, 128, 0, 1,
                sk_b[si][0], nullptr, nullptr, nullptr, 1, BN_SCALE_H);
    launch_gemm(stream, Xin, sk_w[si][1], X2b, 2048, 128, d, d, d, 128, 0, 1,
                sk_b[si][1], nullptr, nullptr, nullptr, 1, BN_SCALE_H);
    launch_gemm(stream, Xin, sk_w[si][2], X3buf, 2048, d, d, d, d, d, 0, 1,
                sk_b[si][2], nullptr, nullptr, nullptr, 1, BN_SCALE_H);
    launch_gemm(stream, X1b, X2b, feat, 2048, 2048, 128, 128, 128, 2048, 0, 1,
                nullptr, nullptr, nullptr, nullptr, 0, 1.f);
    softmax_col_k<<<cdiv(2048, 256), 256, 0, stream>>>(feat, 2048, 2048);
    launch_gemm(stream, feat, X3buf, outbuf, 2048, d, 2048, 2048, d, d, 0, 0,
                nullptr, nullptr, nullptr, Xin, 0, 1.f);
  };

  skip(Bdt, 256, 0, X3s, Xs);
  skip(KBdt, 256, 1, X3s, Ks);

  // ---- euclid distance via fused GEMM epilogue
  rowsq_k<<<8, 256, 0, stream>>>(Xs, xn, 2048, 256);
  rowsq_k<<<8, 256, 0, stream>>>(Ks, yn, 2048, 256);
  launch_gemm(stream, Xs, Ks, aff, 2048, 2048, 256, 256, 256, 2048, 0, 1,
              nullptr, xn, yn, nullptr, 3, 1.f);

  // ---- skip_block(aff) — dominant GEMMs (2048^3)
  skip(aff, 2048, 2, X3big, affs);

  // ---- mask_scores
  launch_gemm(stream, affs, mk_w1, X1b, 2048, 128, 2048, 2048, 2048, 128, 0, 1,
              mk_b1, nullptr, nullptr, nullptr, 1, 1.f);
  launch_gemm(stream, affs, mk_w2, X2b, 2048, 128, 2048, 2048, 2048, 128, 0, 1,
              mk_b2, nullptr, nullptr, nullptr, 2, 1.f);
  launch_gemm(stream, X1b, X2b, Mmat, 128, 128, 2048, 128, 128, 128, 1, 0,
              nullptr, nullptr, nullptr, nullptr, 0, 1.f);
  launch_gemm(stream, Mmat, mk_w3, a3, 128, 2048, 128, 128, 128, 2048, 0, 1,
              mk_b3, nullptr, nullptr, nullptr, 1, 1.f);
  colvec_dot_k<<<8, 256, 0, stream>>>(a3, mk_w4, mk_b4, svec, 2048, 128);
  softmax_vec_k<<<1, 256, 0, stream>>>(svec, scor, 2048);

  // ---- top-S bag + fuse
  topk_k<<<1, 256, 0, stream>>>(scor, idxb, 2048, 204);
  bag_softmax_k<<<2048, 256, 0, stream>>>(affs, idxb, bag, 204);
  concat_bag_k<<<cdiv(2048 * 460, 256), 256, 0, stream>>>(Bdt, bag, catb, 2048, 256, 204);
  launch_gemm(stream, catb, fuse_w, Bfuse, 2048, 256, 460, 460, 460, 256, 0, 1,
              fuse_b, nullptr, nullptr, nullptr, 1, 1.f);

  // ---- attention 1 (din = 256)
  launch_gemm(stream, Bfuse, a1_few, H1, 2048, 256, 256, 256, 256, 256, 0, 1,
              a1_feb, nullptr, nullptr, nullptr, 1, 1.f);
  launch_gemm(stream, H1, a1_vw, X1b, 2048, 128, 256, 256, 256, 128, 0, 1,
              a1_vb, nullptr, nullptr, nullptr, 2, 1.f);
  launch_gemm(stream, H1, a1_uw, X2b, 2048, 128, 256, 256, 256, 128, 0, 1,
              a1_ub, nullptr, nullptr, nullptr, 1, 1.f);
  gate_dot_k<<<8, 256, 0, stream>>>(X1b, X2b, a1_aww, a1_awb, avec, 2048, 128);
  softmax_vec_k<<<1, 256, 0, stream>>>(avec, Avec, 2048);
  wsum_k<<<1, 256, 0, stream>>>(Avec, H1, AH, 2048, 256);
  small_linear_k<<<1, 256, 0, stream>>>(AH, a1_ew, a1_eb, M1v, 256, 256);
  concat_vec_k<<<cdiv(2048 * 512, 256), 256, 0, stream>>>(Bfuse, M1v, X2cat, 2048, 256, 256);

  // ---- attention 2 (din = 512); its softmax A is the second output
  launch_gemm(stream, X2cat, a2_few, H2, 2048, 256, 512, 512, 512, 256, 0, 1,
              a2_feb, nullptr, nullptr, nullptr, 1, 1.f);
  launch_gemm(stream, H2, a2_vw, X1b, 2048, 128, 256, 256, 256, 128, 0, 1,
              a2_vb, nullptr, nullptr, nullptr, 2, 1.f);
  launch_gemm(stream, H2, a2_uw, X2b, 2048, 128, 256, 256, 256, 128, 0, 1,
              a2_ub, nullptr, nullptr, nullptr, 1, 1.f);
  gate_dot_k<<<8, 256, 0, stream>>>(X1b, X2b, a2_aww, a2_awb, avec, 2048, 128);
  softmax_vec_k<<<1, 256, 0, stream>>>(avec, A2v, 2048);
  copy_k<<<8, 256, 0, stream>>>(A2v, out + 1, 2048);           // output A [1,2048]
  wsum_k<<<1, 256, 0, stream>>>(A2v, H2, AH2, 2048, 256);
  small_linear_k<<<2, 256, 0, stream>>>(AH2, a2_ew, a2_eb, m2, 512, 256);
  small_linear_k<<<1, 256, 0, stream>>>(m2, semb_w, semb_b, bvec, 256, 512);
  final_k<<<1, 1, 0, stream>>>(bvec, cls_w, cls_b, out);       // output Y [1,1]
}
